// SessionGraphSoftAttention_33363305955893
// MI455X (gfx1250) — matrix operand order repaired and checked by reading, not compiled, not run on hardware
//
#include <hip/hip_runtime.h>

typedef __attribute__((ext_vector_type(16))) _Float16 v16h;
typedef __attribute__((ext_vector_type(8)))  _Float16 v8h;
typedef __attribute__((ext_vector_type(4)))  _Float16 v4h;
typedef __attribute__((ext_vector_type(4)))  float    v4f;
typedef __attribute__((ext_vector_type(8)))  float    v8f;

constexpr int L    = 50;    // sequence length
constexpr int NN   = 50;    // nodes per batch
constexpr int D    = 128;   // feature dim
constexpr int MPAD = 64;    // L padded to multiple of 16 for WMMA M tiles
constexpr int AST  = 136;   // sA row stride (halves), padded vs 128 to avoid bank conflicts
constexpr int WST  = 136;   // sWt row stride (halves)

union V16 { v16h v; v8h h[2]; };

// Pattern must be a frontend constant -> template parameter.
template <int PAT>
__device__ __forceinline__ float swz_add(float v) {
    return v + __int_as_float(__builtin_amdgcn_ds_swizzle(__float_as_int(v), PAT));
}

__global__ __launch_bounds__(256) void session_attn_wmma(
    const int*   __restrict__ seq_idx,   // [B, L]
    const int*   __restrict__ mask,      // [B, L]
    const float* __restrict__ nodes,     // [B, N, D]
    const float* __restrict__ W_last,    // [D, D] row-major (k, n)
    const float* __restrict__ W_seq,     // [D, D] row-major (k, n)
    const float* __restrict__ b_seq,     // [D]
    const float* __restrict__ W_alpha,   // [D]
    float*       __restrict__ out,       // [B,D] v_n then [B,D] session_graph
    int B)
{
    __shared__ __align__(16) _Float16 sA [MPAD * AST]; // v_i in f16, rows >= L zeroed
    __shared__ __align__(16) _Float16 sWt[D * WST];    // W_seq transposed: [n][k] f16
    __shared__ float sVn [D];
    __shared__ float sVnp[D];     // v_n @ W_last + b_seq
    __shared__ float sWal[D];     // W_alpha
    __shared__ float sAcc[MPAD];  // per-row alpha accumulators
    __shared__ float sAw [L];     // masked alpha
    __shared__ float sMask[L];
    __shared__ int   sIdx[L];
    __shared__ int   sLast;

    const int b    = blockIdx.x;
    const int tid  = threadIdx.x;
    const int lane = tid & 31;
    const int wave = tid >> 5;
    const float* nodeB = nodes + (size_t)b * NN * D;

    // ---- stage 1: indices, mask, small constants ----
    if (tid < L) {
        sIdx[tid]  = seq_idx[b * L + tid];
        sMask[tid] = (float)mask[b * L + tid];
    }
    if (tid < MPAD) sAcc[tid] = 0.0f;
    if (tid < D)    sWal[tid] = W_alpha[tid];
    __syncthreads();

    if (tid == 0) {
        int sl = 0;
        for (int l = 0; l < L; ++l) sl += (int)sMask[l];
        sLast = sIdx[sl > 0 ? sl - 1 : 0];
    }
    __syncthreads();
    const int lastIdx = sLast;

    // ---- stage 2: load v_n (also the first output), gather v_i, stage W_seq^T ----
    if (tid < D) {
        float v = nodeB[lastIdx * D + tid];
        sVn[tid] = v;
        out[(size_t)b * D + tid] = v;               // v_n output (exact f32 copy)
    }
    // gather v_i rows -> f16 A tile (float4 chunks, 50*32 chunks)
    for (int c = tid; c < L * (D / 4); c += 256) {
        const int row = c >> 5;
        const int col = (c & 31) << 2;
        const float4 x = *(const float4*)(nodeB + sIdx[row] * D + col);
        v4h h; h[0] = (_Float16)x.x; h[1] = (_Float16)x.y;
               h[2] = (_Float16)x.z; h[3] = (_Float16)x.w;
        *(v4h*)(sA + row * AST + col) = h;
    }
    // zero-pad rows L..MPAD-1
    for (int c = tid; c < (MPAD - L) * (D / 4); c += 256) {
        const int row = L + (c >> 5);
        const int col = (c & 31) << 2;
        v4h z = {};
        *(v4h*)(sA + row * AST + col) = z;
    }
    // W_seq[k][n] -> sWt[n][k] f16 via 4x4 micro-tiles:
    // 4 coalesced float4 global loads + 4 ds_store_b64 per 16 elements
    for (int t = tid; t < (D / 4) * (D / 4); t += 256) {
        const int k = (t >> 5) << 2;      // k-quad base
        const int n = (t & 31) << 2;      // n-quad base
        v4f r[4];
        #pragma unroll
        for (int j = 0; j < 4; ++j)
            r[j] = *(const v4f*)(W_seq + (k + j) * D + n);
        #pragma unroll
        for (int i = 0; i < 4; ++i) {
            v4h hh;
            hh[0] = (_Float16)r[0][i]; hh[1] = (_Float16)r[1][i];
            hh[2] = (_Float16)r[2][i]; hh[3] = (_Float16)r[3][i];
            *(v4h*)(sWt + (n + i) * WST + k) = hh;
        }
    }
    __syncthreads();

    // ---- stage 3: vn_proj = v_n @ W_last + b_seq (VALU, tiny) ----
    if (tid < D) {
        float acc = 0.0f;
        #pragma unroll 4
        for (int k = 0; k < D; ++k)
            acc = fmaf(sVn[k], W_last[k * D + tid], acc);
        sVnp[tid] = acc + b_seq[tid];
    }
    __syncthreads();

    // ---- stage 4: vi_proj GEMM on WMMA + fused sigmoid/alpha epilogue ----
    {
        const int nt = wave;               // each of 8 waves owns one 16-col tile
        const int hf = (lane >> 4) & 1;    // which half-wave
        const int lr = lane & 15;

        // B fragments for all 4 K-steps (column-contiguous reads from sWt)
        V16 Bf[4];
        #pragma unroll
        for (int kt = 0; kt < 4; ++kt) {
            const int ncol = nt * 16 + lr;
            const int k0   = kt * 32 + hf * 16;
            const _Float16* p = sWt + ncol * WST + k0;
            Bf[kt].h[0] = *(const v8h*)(p);
            Bf[kt].h[1] = *(const v8h*)(p + 8);
        }

        #pragma unroll
        for (int mt = 0; mt < 4; ++mt) {
            v8f Cf = {};
            #pragma unroll
            for (int kt = 0; kt < 4; ++kt) {
                V16 Af;
                const int row = mt * 16 + lr;
                const int c0  = kt * 32 + hf * 16;
                const _Float16* p = sA + row * AST + c0;
                Af.h[0] = *(const v8h*)(p);
                Af.h[1] = *(const v8h*)(p + 8);
                Cf = __builtin_amdgcn_wmma_f32_16x16x32_f16(
                        /*neg_a=*/false, Af.v, /*neg_b=*/false, Bf[kt].v,
                        /*c_mod=*/(short)0, Cf, /*reuse_a=*/false, /*reuse_b=*/false);
            }
            // epilogue: alpha partial = sum_n sigmoid(x + vnp[n]) * W_alpha[n]
            const int   ncol = nt * 16 + lr;
            const float vp   = sVnp[ncol];
            const float wa   = sWal[ncol];

            float part[8];
            #pragma unroll
            for (int r = 0; r < 8; ++r) {
                const float x = Cf[r] + vp;
                part[r] = __builtin_amdgcn_rcpf(1.0f + __expf(-x)) * wa;
            }
            // two butterfly levels, batched across rows for DS-level ILP
            #pragma unroll
            for (int r = 0; r < 8; ++r) part[r] = swz_add<0x201F>(part[r]); // SWAPX8
            #pragma unroll
            for (int r = 0; r < 8; ++r) part[r] = swz_add<0x101F>(part[r]); // SWAPX4
            // lanes lr=0..3 hold disjoint 4-lane partials {l, l^4, l^8, l^12}
            if (lr < 4) {
                const int mbase = mt * 16 + hf * 8;
                #pragma unroll
                for (int r = 0; r < 8; ++r)
                    atomicAdd(&sAcc[mbase + r], part[r]);
            }
        }
    }
    __syncthreads();

    // ---- stage 5: mask alpha, then session_graph = sum_l alpha_l * v_i[l] ----
    if (tid < L) sAw[tid] = sAcc[tid] * sMask[tid];
    __syncthreads();

    if (tid < D) {
        float acc = 0.0f;
        #pragma unroll 2
        for (int l = 0; l < L; ++l)
            acc = fmaf(sAw[l], nodeB[sIdx[l] * D + tid], acc);  // L0/L2 hits
        out[((size_t)B + b) * D + tid] = acc;
    }
}

extern "C" void kernel_launch(void* const* d_in, const int* in_sizes, int n_in,
                              void* d_out, int out_size, void* d_ws, size_t ws_size,
                              hipStream_t stream) {
    const int*   seq_idx = (const int*)d_in[0];
    const int*   mask    = (const int*)d_in[1];
    const float* nodes   = (const float*)d_in[2];
    // d_in[3] = batch_size scalar on device (unused; derived from in_sizes)
    const float* W_last  = (const float*)d_in[4];
    const float* W_seq   = (const float*)d_in[5];
    const float* b_seq   = (const float*)d_in[6];
    const float* W_alpha = (const float*)d_in[7];
    float*       out     = (float*)d_out;

    const int B = in_sizes[0] / L;   // in_sizes[0] == B * L

    hipLaunchKernelGGL(session_attn_wmma, dim3(B), dim3(256), 0, stream,
                       seq_idx, mask, nodes, W_last, W_seq, b_seq, W_alpha, out, B);
}
